// TextEncoder_30580167148130
// MI455X (gfx1250) — matrix operand order repaired and checked by reading, not compiled, not run on hardware
//
#include <hip/hip_runtime.h>
#include <hip/hip_bf16.h>

// ---------------------------------------------------------------------------
// CLIP text encoder forward, CDNA5 (gfx1250), wave32 + WMMA f16->f32.
// GEMMs: v_wmma_f32_16x16x32_f16 with async global->LDS double-buffered
// staging (GLOBAL_LOAD_ASYNC_TO_LDS_B128, ASYNCcnt pipeline).
// ---------------------------------------------------------------------------

typedef __attribute__((ext_vector_type(16))) _Float16 v16h;
typedef __attribute__((ext_vector_type(8)))  float    v8f;

#define LNUM   12
#define DMODEL 512
#define NHEAD  8
#define SEQ    77
#define EMB    512
#define BATCH  512
#define HDIM   64
#define NTOK   (BATCH * SEQ)   // 39424 tokens = 308 * 128

union AFrag { v16h v; _Float16 h[16]; };
union CFrag { v8f  v; float    f[8];  };

// ---- WMMA fragment loaders (contiguous k = 16*(lane/16) + e in A and B, wave32) -------------

// A operand: matrix stored row-major [M][K]; 16x32 tile.
__device__ inline v16h frag_a_rowmajor(const _Float16* base, int row0, int k0, int ld) {
  int lane = threadIdx.x & 31;
  const _Float16* p = base + (size_t)(row0 + (lane & 15)) * ld + k0 + ((lane >> 4) << 4);
  AFrag a;
#pragma unroll
  for (int e = 0; e < 8; ++e)  a.h[e] = p[e];
#pragma unroll
  for (int e = 8; e < 16; ++e) a.h[e] = p[e];
  return a.v;
}

// B operand when the matrix is stored [N][K] row-major (computing A * W^T).
__device__ inline v16h frag_b_nk(const _Float16* base, int n0, int k0, int ld) {
  int lane = threadIdx.x & 31;
  const _Float16* p = base + (size_t)(n0 + (lane & 15)) * ld + k0 + ((lane >> 4) << 4);
  AFrag b;
#pragma unroll
  for (int e = 0; e < 16; ++e) b.h[e] = p[e];
  return b.v;
}

// B operand when matrix stored [K][N] row-major (e.g. V in attention).
__device__ inline v16h frag_b_kn(const _Float16* base, int k0, int n0, int ld) {
  int lane = threadIdx.x & 31;
  int n  = n0 + (lane & 15);
  int kb = k0 + ((lane >> 4) << 4);
  AFrag b;
#pragma unroll
  for (int e = 0; e < 16; ++e) b.h[e] = base[(size_t)(kb + e) * ld + n];
  return b.v;
}

__device__ inline v8f wmma_f16(v16h a, v16h b, v8f c) {
  return __builtin_amdgcn_wmma_f32_16x16x32_f16(false, a, false, b, (short)0, c, false, false);
}

// ---- CDNA5 async global->LDS copy (ASYNCcnt-tracked, no VGPR round trip) ---
__device__ inline void async_b128(unsigned lds_addr, const _Float16* gaddr) {
  asm volatile("global_load_async_to_lds_b128 %0, %1, off"
               :: "v"(lds_addr), "v"(gaddr) : "memory");
}
__device__ inline void wait_async0() {
  asm volatile("s_wait_asynccnt 0x0" ::: "memory");
}

// ---- Utility kernels -------------------------------------------------------

__global__ __launch_bounds__(256) void f32_to_f16_kernel(
    const float* __restrict__ src, _Float16* __restrict__ dst, size_t n) {
  size_t i = (size_t)blockIdx.x * blockDim.x + threadIdx.x;
  if (i < n) dst[i] = (_Float16)src[i];
}

__global__ __launch_bounds__(256) void transpose_f32_to_f16_kernel(
    const float* __restrict__ src, _Float16* __restrict__ dst, int rows, int cols) {
  int i = blockIdx.x * blockDim.x + threadIdx.x;
  if (i < rows * cols) {
    int r = i / cols, c = i % cols;
    dst[(size_t)c * rows + r] = (_Float16)src[i];
  }
}

__global__ __launch_bounds__(256) void add_pos_kernel(
    const float* __restrict__ prompts, const float* __restrict__ pos,
    float* __restrict__ x) {
  size_t i = (size_t)blockIdx.x * blockDim.x + threadIdx.x;
  if (i < (size_t)NTOK * DMODEL) {
    int sd = (int)(i % ((size_t)SEQ * DMODEL));
    x[i] = prompts[i] + pos[sd];
  }
}

// One wave32 per row; 512 elems -> 16 per lane; shfl_xor tree reductions.
__global__ __launch_bounds__(256) void layernorm_kernel(
    const float* __restrict__ x, const float* __restrict__ w,
    const float* __restrict__ b, _Float16* __restrict__ out16,
    float* __restrict__ out32, int M) {
  int row  = blockIdx.x * 8 + (threadIdx.x >> 5);
  int lane = threadIdx.x & 31;
  if (row >= M) return;
  const float* xr = x + (size_t)row * DMODEL;
  float vals[16];
  float s = 0.f;
#pragma unroll
  for (int i = 0; i < 16; ++i) { vals[i] = xr[lane + i * 32]; s += vals[i]; }
#pragma unroll
  for (int off = 16; off >= 1; off >>= 1) s += __shfl_xor(s, off, 32);
  float mean = s * (1.f / (float)DMODEL);
  float vs = 0.f;
#pragma unroll
  for (int i = 0; i < 16; ++i) { float d = vals[i] - mean; vs += d * d; }
#pragma unroll
  for (int off = 16; off >= 1; off >>= 1) vs += __shfl_xor(vs, off, 32);
  float rstd = rsqrtf(vs * (1.f / (float)DMODEL) + 1e-5f);
#pragma unroll
  for (int i = 0; i < 16; ++i) {
    int c = lane + i * 32;
    float y = (vals[i] - mean) * rstd * w[c] + b[c];
    if (out16) out16[(size_t)row * DMODEL + c] = (_Float16)y;
    if (out32) out32[(size_t)row * DMODEL + c] = y;
  }
}

// ---- Tiled WMMA GEMM: C[M,N] = act(A[M,K] * W[N,K]^T + bias + residual) ----
// Block: 256 threads = 8 waves, tile 128x128, wave sub-tile 32x64.
// REQUIRES: M % 128 == 0, N % 128 == 0, K % 32 == 0 (true for every call here)
// -> completely branch-free staging & epilogue.
// Double-buffered LDS fed by GLOBAL_LOAD_ASYNC_TO_LDS_B128; one s_wait_asynccnt
// + one barrier per K-step; next tile's HBM fetch overlaps current WMMAs.
#define GBM 128
#define GBN 128
#define GBK 32

__global__ __launch_bounds__(256) void gemm_f16_kernel(
    const _Float16* __restrict__ A, const _Float16* __restrict__ W,
    const float* __restrict__ bias, const float* residual,
    float* out32, _Float16* out16, int M, int N, int K, int act) {
  __shared__ _Float16 As[2][GBM * GBK];
  __shared__ _Float16 Ws[2][GBN * GBK];
  int tid  = threadIdx.x;
  int wave = tid >> 5, lane = tid & 31;
  int bm = blockIdx.y * GBM, bn = blockIdx.x * GBN;
  int wm = (wave >> 1) * 32;   // 4 waves along M
  int wn = (wave & 1) * 64;    // 2 waves along N

  // Per-thread staging slots: 32B (two b128 chunks) of A and of W per K-step.
  int srow = tid >> 1;             // 0..127
  int scb  = (tid & 1) * 16;       // 0 or 16 halfs
  const _Float16* agp = A + (size_t)(bm + srow) * K + scb;
  const _Float16* wgp = W + (size_t)(bn + srow) * K + scb;
  unsigned alds0 = (unsigned)(size_t)&As[0][srow * GBK + scb];
  unsigned wlds0 = (unsigned)(size_t)&Ws[0][srow * GBK + scb];
  const unsigned abuf = (unsigned)(GBM * GBK * 2);   // bytes per A buffer
  const unsigned wbuf = (unsigned)(GBN * GBK * 2);   // bytes per W buffer

  CFrag acc[2][4] = {};
  int KT = K / GBK;

  // Prefetch tile 0 into buffer 0.
  async_b128(alds0,      agp);
  async_b128(alds0 + 16, agp + 8);
  async_b128(wlds0,      wgp);
  async_b128(wlds0 + 16, wgp + 8);

  for (int kt = 0; kt < KT; ++kt) {
    int cur = kt & 1;
    wait_async0();      // this wave's writes into buf[cur] have landed in LDS
    __syncthreads();    // all waves' writes landed; all done reading buf[cur^1]
    if (kt + 1 < KT) {  // issue next tile into the other buffer (overlaps WMMA)
      int nxt = cur ^ 1;
      const _Float16* ag = agp + (kt + 1) * GBK;
      const _Float16* wg = wgp + (kt + 1) * GBK;
      async_b128(alds0 + nxt * abuf,      ag);
      async_b128(alds0 + nxt * abuf + 16, ag + 8);
      async_b128(wlds0 + nxt * wbuf,      wg);
      async_b128(wlds0 + nxt * wbuf + 16, wg + 8);
    }
    const _Float16* Ab = As[cur];
    const _Float16* Wb = Ws[cur];
    v16h af[2], bf[4];
#pragma unroll
    for (int i = 0; i < 2; ++i) af[i] = frag_a_rowmajor(Ab, wm + i * 16, 0, GBK);
#pragma unroll
    for (int j = 0; j < 4; ++j) bf[j] = frag_b_nk(Wb, wn + j * 16, 0, GBK);
#pragma unroll
    for (int i = 0; i < 2; ++i)
#pragma unroll
      for (int j = 0; j < 4; ++j)
        acc[i][j].v = wmma_f16(af[i], bf[j], acc[i][j].v);
  }

  // Epilogue (branch-free indexing; uniform feature flags).
#pragma unroll
  for (int i2 = 0; i2 < 2; ++i2)
#pragma unroll
    for (int j2 = 0; j2 < 4; ++j2) {
      int col = bn + wn + j2 * 16 + (lane & 15);
#pragma unroll
      for (int e = 0; e < 8; ++e) {
        int row = bm + wm + i2 * 16 + e + ((lane >> 4) << 3);
        float v = acc[i2][j2].f[e] + (bias ? bias[col] : 0.f);
        if (residual) v += residual[(size_t)row * N + col];
        if (act == 1) v = v / (1.f + __expf(-1.702f * v));  // quickGELU
        if (out32) out32[(size_t)row * N + col] = v;
        if (out16) out16[(size_t)row * N + col] = (_Float16)v;
      }
    }
}

// ---- Attention: one workgroup per (b,h). S padded 77->80(scores)/96(K of AV).
__global__ __launch_bounds__(256) void attention_kernel(
    const _Float16* __restrict__ qkv,   // [NTOK][1536] : Q|K|V
    _Float16* __restrict__ out) {       // [NTOK][512]
  __shared__ __align__(16) char smem[61440];
  float*    Ss = (float*)smem;                   // 80*80 f32  (25600 B)
  _Float16* Ps = (_Float16*)(smem + 25600);      // 80*96 f16  (15360 B)
  _Float16* Qs = (_Float16*)(smem + 40960);      // 80*64 f16  (10240 B)
  _Float16* Ks = (_Float16*)(smem + 51200);      // 80*64 f16  (10240 B)
  _Float16* Vs = (_Float16*)(smem + 40960);      // 96*64 f16, reuses Q/K space

  int b = blockIdx.x >> 3;
  int h = blockIdx.x & 7;
  int tid  = threadIdx.x;
  int wave = tid >> 5, lane = tid & 31;
  size_t base = (size_t)b * SEQ * 1536 + h * HDIM;

  for (int idx = tid; idx < 80 * 64; idx += 256) {
    int s = idx >> 6, d = idx & 63;
    _Float16 q = (_Float16)0, k = (_Float16)0;
    if (s < SEQ) {
      q = qkv[base + (size_t)s * 1536 + d];
      k = qkv[base + (size_t)s * 1536 + 512 + d];
    }
    Qs[idx] = q; Ks[idx] = k;
  }
  __syncthreads();

  // Scores: 5x5 tiles of 16x16, K = 64 (two WMMA steps).
  for (int t = wave; t < 25; t += 8) {
    int ti = t / 5, tj = t % 5;
    CFrag acc = {};
    for (int k0 = 0; k0 < 64; k0 += 32) {
      v16h a  = frag_a_rowmajor(Qs, ti * 16, k0, 64);
      v16h bf = frag_b_nk(Ks, tj * 16, k0, 64);
      acc.v = wmma_f16(a, bf, acc.v);
    }
    int col = tj * 16 + (lane & 15);
#pragma unroll
    for (int e = 0; e < 8; ++e) {
      int row = ti * 16 + e + ((lane >> 4) << 3);
      float v = acc.f[e] * 0.125f;                 // 1/sqrt(64)
      if (col > row || col >= SEQ) v = -1e4f;      // CLIP causal additive mask
      Ss[row * 80 + col] = v;
    }
  }
  __syncthreads();

  for (int idx = tid; idx < 96 * 64; idx += 256) {
    int s = idx >> 6, d = idx & 63;
    Vs[idx] = (s < SEQ) ? qkv[base + (size_t)s * 1536 + 1024 + d] : (_Float16)0;
  }
  if (tid < 80) {
    int r = tid;
    if (r < SEQ) {
      float mx = -1e30f;
      for (int c = 0; c < SEQ; ++c) mx = fmaxf(mx, Ss[r * 80 + c]);
      float sum = 0.f;
      for (int c = 0; c < SEQ; ++c) { float e = __expf(Ss[r * 80 + c] - mx); Ss[r * 80 + c] = e; sum += e; }
      float inv = 1.f / sum;
      for (int c = 0; c < 96; ++c)
        Ps[r * 96 + c] = (c < SEQ) ? (_Float16)(Ss[r * 80 + c] * inv) : (_Float16)0;
    } else {
      for (int c = 0; c < 96; ++c) Ps[r * 96 + c] = (_Float16)0;
    }
  }
  __syncthreads();

  // O = P @ V : 5x4 output tiles, K = 96 (three WMMA steps).
  for (int t = wave; t < 20; t += 8) {
    int ti = t / 4, tj = t % 4;
    CFrag acc = {};
    for (int k0 = 0; k0 < 96; k0 += 32) {
      v16h a  = frag_a_rowmajor(Ps, ti * 16, k0, 96);
      v16h bf = frag_b_kn(Vs, k0, tj * 16, 64);
      acc.v = wmma_f16(a, bf, acc.v);
    }
    int col = tj * 16 + (lane & 15);
#pragma unroll
    for (int e = 0; e < 8; ++e) {
      int row = ti * 16 + e + ((lane >> 4) << 3);
      if (row < SEQ)
        out[((size_t)(b * SEQ + row)) * DMODEL + h * HDIM + col] = (_Float16)acc.f[e];
    }
  }
}

// ---- Ragged mean-pool over [0, eot] ----------------------------------------
__global__ __launch_bounds__(256) void pool_kernel(
    const float* __restrict__ xln, const long long* __restrict__ toks,
    float* __restrict__ pooled32, _Float16* __restrict__ pooled16) {
  int b = blockIdx.x;
  __shared__ int eot_s;
  if (threadIdx.x == 0) {
    long long best = toks[(size_t)b * SEQ];
    int bi = 0;
    for (int s = 1; s < SEQ; ++s) {
      long long v = toks[(size_t)b * SEQ + s];
      if (v > best) { best = v; bi = s; }     // first max, like jnp.argmax
    }
    eot_s = bi;
  }
  __syncthreads();
  int eot = eot_s;
  for (int d = threadIdx.x; d < DMODEL; d += blockDim.x) {
    float s = 0.f;
    for (int t = 0; t <= eot; ++t) s += xln[((size_t)b * SEQ + t) * DMODEL + d];
    float p = s / (float)(eot + 1);
    pooled32[(size_t)b * DMODEL + d] = p;
    pooled16[(size_t)b * DMODEL + d] = (_Float16)p;
  }
}

// ---------------------------------------------------------------------------

extern "C" void kernel_launch(void* const* d_in, const int* in_sizes, int n_in,
                              void* d_out, int out_size, void* d_ws, size_t ws_size,
                              hipStream_t stream) {
  (void)in_sizes; (void)n_in; (void)out_size; (void)ws_size;

  const float*      prompts = (const float*)d_in[0];
  const long long*  toks    = (const long long*)d_in[1];
  const float*      pos_emb = (const float*)d_in[2];
  const float*      ln1_w   = (const float*)d_in[3];
  const float*      ln1_b   = (const float*)d_in[4];
  const float*      qkv_w   = (const float*)d_in[5];
  const float*      qkv_b   = (const float*)d_in[6];
  const float*      proj_w  = (const float*)d_in[7];
  const float*      proj_b  = (const float*)d_in[8];
  const float*      ln2_w   = (const float*)d_in[9];
  const float*      ln2_b   = (const float*)d_in[10];
  const float*      fc1_w   = (const float*)d_in[11];
  const float*      fc1_b   = (const float*)d_in[12];
  const float*      fc2_w   = (const float*)d_in[13];
  const float*      fc2_b   = (const float*)d_in[14];
  const float*      lnf_w   = (const float*)d_in[15];
  const float*      lnf_b   = (const float*)d_in[16];
  const float*      tproj   = (const float*)d_in[17];

  char* ws = (char*)d_ws;
  size_t off = 0;
  auto take = [&](size_t bytes) -> char* {
    char* p = ws + off;
    off += (bytes + 255) & ~(size_t)255;
    return p;
  };
  float*     xf       = (float*)take((size_t)NTOK * DMODEL * 4);
  _Float16*  hbig     = (_Float16*)take((size_t)NTOK * 2048 * 2);
  _Float16*  hsmall   = (_Float16*)take((size_t)NTOK * DMODEL * 2);
  float*     pooled32 = (float*)take((size_t)BATCH * DMODEL * 4);
  _Float16*  pooled16 = (_Float16*)take((size_t)BATCH * DMODEL * 2);
  _Float16*  qkvw16   = (_Float16*)take((size_t)LNUM * 3 * DMODEL * DMODEL * 2);
  _Float16*  projw16  = (_Float16*)take((size_t)LNUM * DMODEL * DMODEL * 2);
  _Float16*  fc1w16   = (_Float16*)take((size_t)LNUM * 4 * DMODEL * DMODEL * 2);
  _Float16*  fc2w16   = (_Float16*)take((size_t)LNUM * DMODEL * 4 * DMODEL * 2);
  _Float16*  tprojT16 = (_Float16*)take((size_t)EMB * DMODEL * 2);

  auto cvt = [&](const float* s, _Float16* d, size_t n) {
    f32_to_f16_kernel<<<(unsigned)((n + 255) / 256), 256, 0, stream>>>(s, d, n);
  };
  cvt(qkv_w,  qkvw16,  (size_t)LNUM * 3 * DMODEL * DMODEL);
  cvt(proj_w, projw16, (size_t)LNUM * DMODEL * DMODEL);
  cvt(fc1_w,  fc1w16,  (size_t)LNUM * 4 * DMODEL * DMODEL);
  cvt(fc2_w,  fc2w16,  (size_t)LNUM * DMODEL * 4 * DMODEL);
  transpose_f32_to_f16_kernel<<<(DMODEL * EMB + 255) / 256, 256, 0, stream>>>(
      tproj, tprojT16, DMODEL, EMB);

  add_pos_kernel<<<(unsigned)(((size_t)NTOK * DMODEL + 255) / 256), 256, 0, stream>>>(
      prompts, pos_emb, xf);

  dim3 gM(DMODEL / GBN, NTOK / GBM);           // N=512
  dim3 gQKV((3 * DMODEL) / GBN, NTOK / GBM);   // N=1536
  dim3 gFC1((4 * DMODEL) / GBN, NTOK / GBM);   // N=2048
  unsigned lnBlocks = (NTOK + 7) / 8;

  for (int l = 0; l < LNUM; ++l) {
    layernorm_kernel<<<lnBlocks, 256, 0, stream>>>(
        xf, ln1_w + l * DMODEL, ln1_b + l * DMODEL, hsmall, nullptr, NTOK);
    gemm_f16_kernel<<<gQKV, 256, 0, stream>>>(
        hsmall, qkvw16 + (size_t)l * 3 * DMODEL * DMODEL, qkv_b + l * 3 * DMODEL,
        nullptr, nullptr, hbig, NTOK, 3 * DMODEL, DMODEL, 0);
    attention_kernel<<<BATCH * NHEAD, 256, 0, stream>>>(hbig, hsmall);
    gemm_f16_kernel<<<gM, 256, 0, stream>>>(
        hsmall, projw16 + (size_t)l * DMODEL * DMODEL, proj_b + l * DMODEL,
        xf, xf, nullptr, NTOK, DMODEL, DMODEL, 0);
    layernorm_kernel<<<lnBlocks, 256, 0, stream>>>(
        xf, ln2_w + l * DMODEL, ln2_b + l * DMODEL, hsmall, nullptr, NTOK);
    gemm_f16_kernel<<<gFC1, 256, 0, stream>>>(
        hsmall, fc1w16 + (size_t)l * 4 * DMODEL * DMODEL, fc1_b + l * 4 * DMODEL,
        nullptr, nullptr, hbig, NTOK, 4 * DMODEL, DMODEL, 1);
    gemm_f16_kernel<<<gM, 256, 0, stream>>>(
        hbig, fc2w16 + (size_t)l * DMODEL * 4 * DMODEL, fc2_b + l * DMODEL,
        xf, xf, nullptr, NTOK, DMODEL, 4 * DMODEL, 0);
  }

  float* lnf32 = (float*)hbig;
  layernorm_kernel<<<lnBlocks, 256, 0, stream>>>(xf, lnf_w, lnf_b, nullptr, lnf32, NTOK);
  pool_kernel<<<BATCH, 256, 0, stream>>>(lnf32, toks, pooled32, pooled16);
  dim3 gOut(EMB / GBN, BATCH / GBM);
  gemm_f16_kernel<<<gOut, 256, 0, stream>>>(
      pooled16, tprojT16, nullptr, nullptr, (float*)d_out, nullptr,
      BATCH, EMB, DMODEL, 0);
}
